// OrderedMemoryRecurrent_21921513079034
// MI455X (gfx1250) — matrix operand order repaired and checked by reading, not compiled, not run on hardware
//
#include <hip/hip_runtime.h>
#include <hip/hip_bf16.h>
#include <math.h>

typedef __attribute__((ext_vector_type(16))) _Float16 v16h;
typedef __attribute__((ext_vector_type(8)))  _Float16 v8h;
typedef __attribute__((ext_vector_type(8)))  float    v8f;

namespace {

constexpr int kT = 128, kB = 64, kIN = 512, kH = 256, kNS = 8;
constexpr int kH4 = 1024;
constexpr float kLnEps = 1e-5f;
constexpr float kInvSqrtH = 0.0625f;   // 1/sqrt(256)

// d_out layout (floats): [Xp | memories | probs]
constexpr size_t kMemOff  = (size_t)kT * kB * kH;                   // 2,097,152
constexpr size_t kProbOff = kMemOff + (size_t)kT * kB * kNS * kH;   // 18,874,368

// padded f16 LDS row strides (in halfs) to spread banks; rows stay 16B aligned
constexpr int kSA1 = 520;    // concat [h | Mn] rows: 512 cols + 8 pad
constexpr int kSAS = 264;    // cM slot staging: 256 + 8
constexpr int kSZ1 = 1032;   // relu(z1): 1024 + 8

// workspace layout (bytes): f16 transposed weights + per-WG M state
constexpr size_t kWsWprojT = 0;                         // [256][512] f16
constexpr size_t kWsWqT    = kWsWprojT + 256*512*2;     // [256][256] f16
constexpr size_t kWsWkT    = kWsWqT    + 256*256*2;     // [256][256] f16
constexpr size_t kWsW1T    = kWsWkT    + 256*256*2;     // [1024][512] f16
constexpr size_t kWsW2T    = kWsW1T    + 1024*512*2;    // [1024][1024] f16
constexpr size_t kWsM      = kWsW2T    + 1024*1024*2;   // [4][16][8][256] f32

// scan kernel dynamic LDS layout (bytes)  -> total ~283 KB of 320 KB/WGP
constexpr size_t kLA1  = 0;                       // f16 16 x kSA1
constexpr size_t kLAS  = kLA1  + 16*kSA1*2;       // f16 16 x kSAS
constexpr size_t kLZ1  = kLAS  + 16*kSAS*2;       // f16 16 x kSZ1
constexpr size_t kLZ   = kLZ1  + 16*kSZ1*2;       // f32 16 x 1024 (also k-buffer)
constexpr size_t kLQ   = kLZ   + 16*1024*4;       // f32 16 x 256
constexpr size_t kLIn  = kLQ   + 16*256*4;        // f32 16 x 256
constexpr size_t kLCM  = kLIn  + 16*256*4;        // f32 16 x 8 x 256
constexpr size_t kLP   = kLCM  + 16*8*256*4;      // f32 16 x 8 (prev p)
constexpr size_t kLCP  = kLP   + 16*8*4;
constexpr size_t kLRCP = kLCP  + 16*8*4;
constexpr size_t kLBeta= kLRCP + 16*8*4;
constexpr size_t kLMean= kLBeta+ 16*8*4;          // f32 16
constexpr size_t kLRstd= kLMean+ 16*4;
constexpr size_t kLPad = kLRstd+ 16*4;            // int 16
constexpr size_t kScanLds = kLPad + 16*4;

// ---- WMMA fragment helpers (layouts per CDNA5 ISA 7.12.2) -----------------

// A 16x32 f16, row-major in LDS with row stride strideH halfs.
// lane<16: K 0..7 (frag[0..7]) and 16..23 (frag[8..15]); lane>=16: +8.
static __device__ __forceinline__ v16h frag_a(const _Float16* A, int strideH,
                                              int kk, int lane) {
  int m = lane & 15, hb = (lane >> 4) & 1;
  const _Float16* p = A + m * strideH + kk + hb * 8;
  v8h lo = *(const v8h*)(p);
  v8h hi = *(const v8h*)(p + 16);
  v16h a;
#pragma unroll
  for (int j = 0; j < 8; ++j) { a[j] = lo[j]; a[j + 8] = hi[j]; }
  return a;
}

// B 32x16 f16 from pre-transposed global weight WT[N][K].
// lane<16: K 0..7 and 16..23, lane>=16: +8, column n = n0 + (lane&15).
static __device__ __forceinline__ v16h frag_b(const _Float16* WT, int K,
                                              int n0, int kk, int lane) {
  int n = n0 + (lane & 15), hb = (lane >> 4) & 1;
  const _Float16* p = WT + (size_t)n * K + kk + hb * 8;
  v8h lo = *(const v8h*)(p);
  v8h hi = *(const v8h*)(p + 16);
  v16h b;
#pragma unroll
  for (int j = 0; j < 8; ++j) { b[j] = lo[j]; b[j + 8] = hi[j]; }
  return b;
}

static __device__ __forceinline__ v8f wmma32(v16h a, v16h b, v8f c) {
  return __builtin_amdgcn_wmma_f32_16x16x32_f16(false, a, false, b,
                                                (short)0, c, false, false);
}

static __device__ __forceinline__ float wave_sum(float v) {
  v += __shfl_xor(v, 16); v += __shfl_xor(v, 8); v += __shfl_xor(v, 4);
  v += __shfl_xor(v, 2);  v += __shfl_xor(v, 1);
  return v;
}

} // namespace

// ---- Kernel 1: f32 -> f16 transpose-convert of a weight ---------------------
__global__ void omr_convt_kernel(const float* __restrict__ in,
                                 _Float16* __restrict__ out, int K, int N) {
  int i = blockIdx.x * blockDim.x + threadIdx.x;
  if (i < K * N) {
    int n = i / K;
    int k = i - n * K;
    out[i] = (_Float16)in[(size_t)k * N + n];
  }
}

// ---- Kernel 2: Xp = tanh(LN(X @ W_proj + b_proj)) ---------------------------
__global__ __launch_bounds__(512)
void omr_proj_kernel(const float* __restrict__ X,
                     const _Float16* __restrict__ WpT,
                     const float* __restrict__ bp,
                     const float* __restrict__ lns,
                     const float* __restrict__ lnb,
                     float* __restrict__ out) {
  __shared__ _Float16 sX[16 * kSA1];
  __shared__ float sO[16 * kH];
  const int tid = threadIdx.x, lane = tid & 31, wv = tid >> 5;
  const int r0 = blockIdx.x * 16;

  for (int k = tid; k < 16 * kIN; k += 512) {
    int m = k >> 9, j = k & 511;
    sX[m * kSA1 + j] = (_Float16)X[(size_t)(r0 + m) * kIN + j];
  }
  __syncthreads();

  v8f acc = {0.f, 0.f, 0.f, 0.f, 0.f, 0.f, 0.f, 0.f};
  const int n0 = wv * 16;
  for (int kk = 0; kk < kIN; kk += 32) {
    v16h a = frag_a(sX, kSA1, kk, lane);
    v16h b = frag_b(WpT, kIN, n0, kk, lane);
    acc = wmma32(a, b, acc);
  }
  const int col = n0 + (lane & 15), rb = (lane >> 4) * 8;
#pragma unroll
  for (int r = 0; r < 8; ++r) sO[(rb + r) * kH + col] = acc[r] + bp[col];
  __syncthreads();

  { // wave wv owns row wv: LayerNorm + tanh, write to global
    int m = wv; float s1 = 0.f, s2 = 0.f;
    for (int n = lane; n < kH; n += 32) { float v = sO[m * kH + n]; s1 += v; s2 += v * v; }
    s1 = wave_sum(s1); s2 = wave_sum(s2);
    float mean = s1 * (1.f / kH);
    float var  = s2 * (1.f / kH) - mean * mean;
    float rstd = rsqrtf(var + kLnEps);
    for (int n = lane; n < kH; n += 32) {
      float v = sO[m * kH + n];
      out[(size_t)(r0 + m) * kH + n] = tanhf((v - mean) * rstd * lns[n] + lnb[n]);
    }
  }
}

// ---- Kernel 3: the full recurrent scan. grid=4 (batch row-tiles), 512 thr ---
__global__ __launch_bounds__(512)
void omr_scan_kernel(const float* __restrict__ Xp,     // == d_out base
                     float* __restrict__ out,          // d_out base
                     const unsigned char* __restrict__ mask,
                     const _Float16* __restrict__ WqT,
                     const _Float16* __restrict__ WkT,
                     const _Float16* __restrict__ W1T,
                     const _Float16* __restrict__ W2T,
                     const float* __restrict__ b_q,
                     const float* __restrict__ lnq_s, const float* __restrict__ lnq_b,
                     const float* __restrict__ b_k,
                     const float* __restrict__ lnk_s, const float* __restrict__ lnk_b,
                     const float* __restrict__ W_beta, const float* __restrict__ b_beta,
                     const float* __restrict__ b1, const float* __restrict__ b2,
                     const float* __restrict__ ln_s, const float* __restrict__ ln_b,
                     float* __restrict__ Mstate) {
  extern __shared__ char smem[];
  _Float16* sA1 = (_Float16*)(smem + kLA1);
  _Float16* sAs = (_Float16*)(smem + kLAS);
  _Float16* sZ1 = (_Float16*)(smem + kLZ1);
  float* sZ   = (float*)(smem + kLZ);
  float* sQ   = (float*)(smem + kLQ);
  float* sIn  = (float*)(smem + kLIn);
  float* sCM  = (float*)(smem + kLCM);
  float* sP   = (float*)(smem + kLP);
  float* sCP  = (float*)(smem + kLCP);
  float* sRCP = (float*)(smem + kLRCP);
  float* sBeta= (float*)(smem + kLBeta);
  float* sMean= (float*)(smem + kLMean);
  float* sRstd= (float*)(smem + kLRstd);
  int*   sPad = (int*)(smem + kLPad);

  const int tid = threadIdx.x, lane = tid & 31, wv = tid >> 5;
  const int wg = blockIdx.x;
  const int r0 = wg * 16;                        // batch rows owned
  float* Mg = Mstate + (size_t)wg * 16 * kNS * kH;
  float* memOut  = out + kMemOff;
  float* probOut = out + kProbOff;
  const int rb = (lane >> 4) * 8;

  // ---------------- t = 0 init: M=0, cM[s]=Xp0, p=0 -------------------------
  if (tid < 16) sPad[tid] = mask[r0 + tid] ? 0 : 1;
  for (int k = tid; k < 16 * kH; k += 512) {
    int m = k >> 8, j = k & 255;
    float v = Xp[(size_t)(r0 + m) * kH + j];
    sIn[m * kH + j] = v;
    sA1[m * kSA1 + j] = (_Float16)v;
  }
  __syncthreads();
  for (int k = tid; k < 16 * kNS * kH; k += 512) {
    int m = k >> 11, s = (k >> 8) & 7, j = k & 255;
    float val = sPad[m] ? 0.f : sIn[m * kH + j];
    sCM[k] = val;
    memOut[((size_t)(0 * kB + r0 + m) * kNS + s) * kH + j] = val;
    Mg[k] = 0.f;
  }
  if (tid < 128) {
    int m = tid >> 3, s = tid & 7;
    sP[m * kNS + s] = 0.f;
    probOut[(size_t)(0 * kB + r0 + m) * kNS + s] = 0.f;
  }
  __syncthreads();

  // ---------------- scan t = 1..T-1 -----------------------------------------
  for (int t = 1; t < kT; ++t) {
    if (tid < 16) sPad[tid] = mask[t * kB + r0 + tid] ? 0 : 1;
    for (int k = tid; k < 16 * kH; k += 512) {
      int m = k >> 8, j = k & 255;
      float v = Xp[((size_t)t * kB + r0 + m) * kH + j];
      sIn[m * kH + j] = v;
      sA1[m * kSA1 + j] = (_Float16)v;
    }
    __syncthreads();

    // ---- q = LN(in @ W_q + b_q) : wave wv owns col tile wv ----
    {
      v8f acc = {0.f, 0.f, 0.f, 0.f, 0.f, 0.f, 0.f, 0.f};
      const int n0 = wv * 16;
      for (int kk = 0; kk < kH; kk += 32) {
        v16h a = frag_a(sA1, kSA1, kk, lane);
        v16h b = frag_b(WqT, kH, n0, kk, lane);
        acc = wmma32(a, b, acc);
      }
      const int col = n0 + (lane & 15);
#pragma unroll
      for (int r = 0; r < 8; ++r) sQ[(rb + r) * kH + col] = acc[r] + b_q[col];
    }
    __syncthreads();
    { // q LayerNorm, wave per row
      int m = wv; float s1 = 0.f, s2 = 0.f;
      for (int n = lane; n < kH; n += 32) { float v = sQ[m * kH + n]; s1 += v; s2 += v * v; }
      s1 = wave_sum(s1); s2 = wave_sum(s2);
      float mean = s1 * (1.f / kH), var = s2 * (1.f / kH) - mean * mean;
      float rstd = rsqrtf(var + kLnEps);
      for (int n = lane; n < kH; n += 32) {
        float v = sQ[m * kH + n];
        sQ[m * kH + n] = (v - mean) * rstd * lnq_s[n] + lnq_b[n];
      }
    }
    __syncthreads();

    // ---- per-slot k = LN(cM_s @ W_k), beta_s = relu(q+k)@W_beta ----
    for (int s = 0; s < kNS; ++s) {
      for (int k = tid; k < 16 * kH; k += 512) {
        int m = k >> 8, j = k & 255;
        sAs[m * kSAS + j] = (_Float16)sCM[(m * kNS + s) * kH + j];
      }
      __syncthreads();
      {
        v8f acc = {0.f, 0.f, 0.f, 0.f, 0.f, 0.f, 0.f, 0.f};
        const int n0 = wv * 16;
        for (int kk = 0; kk < kH; kk += 32) {
          v16h a = frag_a(sAs, kSAS, kk, lane);
          v16h b = frag_b(WkT, kH, n0, kk, lane);
          acc = wmma32(a, b, acc);
        }
        const int col = n0 + (lane & 15);
#pragma unroll
        for (int r = 0; r < 8; ++r) sZ[(rb + r) * kH + col] = acc[r] + b_k[col];
      }
      __syncthreads();
      { // k LN stats + fused beta dot-product, wave per row
        int m = wv; float s1 = 0.f, s2 = 0.f;
        for (int n = lane; n < kH; n += 32) { float v = sZ[m * kH + n]; s1 += v; s2 += v * v; }
        s1 = wave_sum(s1); s2 = wave_sum(s2);
        float mean = s1 * (1.f / kH), var = s2 * (1.f / kH) - mean * mean;
        float rstd = rsqrtf(var + kLnEps);
        float dot = 0.f;
        for (int n = lane; n < kH; n += 32) {
          float kn = (sZ[m * kH + n] - mean) * rstd * lnk_s[n] + lnk_b[n];
          float rel = sQ[m * kH + n] + kn;
          rel = rel > 0.f ? rel : 0.f;
          dot += rel * W_beta[n];
        }
        dot = wave_sum(dot);
        if (lane == 0) sBeta[m * kNS + s] = (dot + b_beta[0]) * kInvSqrtH;
      }
      __syncthreads();
    }

    // ---- masked softmax over slots + cumsums (one thread per batch row) ----
    if (tid < 16) {
      const int m = tid;
      float x[kNS], prevcp[kNS], pv[kNS];
      float bmax = -1e30f;
#pragma unroll
      for (int s = 0; s < kNS; ++s) { float b = sBeta[m * kNS + s]; bmax = fmaxf(bmax, b); }
      float c = 0.f;
#pragma unroll
      for (int s = 0; s < kNS; ++s) { c += sP[m * kNS + s]; prevcp[s] = c; }
#pragma unroll
      for (int s = 0; s < kNS; ++s) x[s] = expf(sBeta[m * kNS + s] - bmax);
      float ssum = 0.f;
#pragma unroll
      for (int s = 0; s < kNS; ++s) {
        float mk = (s < kNS - 1) ? ((prevcp[s + 1] < 1e-5f) ? 0.f : prevcp[s + 1]) : 1.f;
        x[s] *= mk;
        ssum += fabsf(x[s]);
      }
      float denom = fmaxf(ssum, 1e-12f);
#pragma unroll
      for (int s = 0; s < kNS; ++s) pv[s] = x[s] / denom;
      float cp = 0.f;
#pragma unroll
      for (int s = 0; s < kNS; ++s) { cp += pv[s]; sCP[m * kNS + s] = cp; }
      float rc = 0.f;
#pragma unroll
      for (int s = kNS - 1; s >= 0; --s) { rc += pv[s]; sRCP[m * kNS + s] = rc; }
      const float padm = sPad[m] ? 0.f : 1.f;
#pragma unroll
      for (int s = 0; s < kNS; ++s) {
        float p = pv[s] * padm;
        sP[m * kNS + s] = p;
        probOut[((size_t)t * kB + r0 + m) * kNS + s] = p;
      }
    }
    __syncthreads();

    // ---- 8 sequential cells ----
    for (int i = 0; i < kNS; ++i) {
      // Mn_i = M_i*(1-rcp_i) + cM_i*rcp_i  ->  A-concat second half + M carry
      for (int k = tid; k < 16 * kH; k += 512) {
        int m = k >> 8, j = k & 255;
        float rc = sRCP[m * kNS + i];
        float mv = Mg[(m * kNS + i) * kH + j];
        float cv = sCM[(m * kNS + i) * kH + j];
        float mn = mv * (1.f - rc) + cv * rc;
        sA1[m * kSA1 + kH + j] = (_Float16)mn;
        Mg[(m * kNS + i) * kH + j] = mn;
      }
      __syncthreads();

      // GEMM1: z1 = relu([h|Mn_i] @ W1 + b1) ; wave owns cols [64wv,64wv+64)
      {
        v8f acc[4];
#pragma unroll
        for (int t4 = 0; t4 < 4; ++t4)
#pragma unroll
          for (int r = 0; r < 8; ++r) acc[t4][r] = 0.f;
        for (int kk = 0; kk < 2 * kH; kk += 32) {
          v16h a = frag_a(sA1, kSA1, kk, lane);
#pragma unroll
          for (int t4 = 0; t4 < 4; ++t4) {
            v16h b = frag_b(W1T, 2 * kH, wv * 64 + t4 * 16, kk, lane);
            acc[t4] = wmma32(a, b, acc[t4]);
          }
        }
#pragma unroll
        for (int t4 = 0; t4 < 4; ++t4) {
          int col = wv * 64 + t4 * 16 + (lane & 15);
#pragma unroll
          for (int r = 0; r < 8; ++r) {
            float v = acc[t4][r] + b1[col];
            v = v > 0.f ? v : 0.f;
            sZ1[(rb + r) * kSZ1 + col] = (_Float16)v;
          }
        }
      }
      __syncthreads();

      // GEMM2: z = z1 @ W2 + b2 ; wave owns gate tiles {3wv,3wv+1,3wv+2} + c tile 48+wv
      {
        const int n0s[4] = {48 * wv, 48 * wv + 16, 48 * wv + 32, 768 + 16 * wv};
#pragma unroll
        for (int t4 = 0; t4 < 4; ++t4)
          __builtin_prefetch(W2T + (size_t)n0s[t4] * kH4, 0, 1);
        v8f acc[4];
#pragma unroll
        for (int t4 = 0; t4 < 4; ++t4)
#pragma unroll
          for (int r = 0; r < 8; ++r) acc[t4][r] = 0.f;
        for (int kk = 0; kk < kH4; kk += 32) {
          v16h a = frag_a(sZ1, kSZ1, kk, lane);
#pragma unroll
          for (int t4 = 0; t4 < 4; ++t4) {
            v16h b = frag_b(W2T, kH4, n0s[t4], kk, lane);
            acc[t4] = wmma32(a, b, acc[t4]);
          }
        }
#pragma unroll
        for (int t4 = 0; t4 < 4; ++t4) {
          int col = n0s[t4] + (lane & 15);
#pragma unroll
          for (int r = 0; r < 8; ++r) sZ[(rb + r) * kH4 + col] = acc[t4][r] + b2[col];
        }
      }
      __syncthreads();

      { // LN stats of c-block (cols 768..1023), wave per row
        int m = wv; float s1 = 0.f, s2 = 0.f;
        for (int n = lane; n < kH; n += 32) { float v = sZ[m * kH4 + 768 + n]; s1 += v; s2 += v * v; }
        s1 = wave_sum(s1); s2 = wave_sum(s2);
        float mean = s1 * (1.f / kH), var = s2 * (1.f / kH) - mean * mean;
        if (lane == 0) { sMean[m] = mean; sRstd[m] = rsqrtf(var + kLnEps); }
      }
      __syncthreads();

      // gating + h update + cM[i] + memories write
      for (int k = tid; k < 16 * kH; k += 512) {
        int m = k >> 8, j = k & 255;
        float z0 = sZ[m * kH4 + 3 * j];
        float z1v = sZ[m * kH4 + 3 * j + 1];
        float z2 = sZ[m * kH4 + 3 * j + 2];
        float mx = fmaxf(z0, fmaxf(z1v, z2));
        float e0 = expf(z0 - mx), e1 = expf(z1v - mx), e2 = expf(z2 - mx);
        float inv = 1.f / (e0 + e1 + e2);
        float cr = sZ[m * kH4 + 768 + j];
        float cn = (cr - sMean[m]) * sRstd[m] * ln_s[j] + ln_b[j];
        float av = tanhf(cn);
        float vi = (float)sA1[m * kSA1 + j];
        float hi_ = (float)sA1[m * kSA1 + kH + j];
        float hc = (e0 * vi + e1 * hi_ + e2 * av) * inv;
        float cpi = sCP[m * kNS + i];
        float hn = sIn[m * kH + j] * (1.f - cpi) + hc * cpi;
        float hout = sPad[m] ? 0.f : hn;
        sA1[m * kSA1 + j] = (_Float16)hn;   // next cell's vi (unpadded chain)
        sCM[(m * kNS + i) * kH + j] = hout;
        memOut[(((size_t)t * kB + r0 + m) * kNS + i) * kH + j] = hout;
      }
      __syncthreads();
    }

    // apply_pad to carried M
    for (int k = tid; k < 16 * kNS * kH; k += 512) {
      int m = k >> 11;
      if (sPad[m]) Mg[k] = 0.f;
    }
    __syncthreads();
  }
}

extern "C" void kernel_launch(void* const* d_in, const int* in_sizes, int n_in,
                              void* d_out, int out_size, void* d_ws, size_t ws_size,
                              hipStream_t stream) {
  (void)in_sizes; (void)n_in; (void)out_size; (void)ws_size;

  const float* X      = (const float*)d_in[0];
  const unsigned char* mask = (const unsigned char*)d_in[1]; // jnp bool_: 1 byte
  // d_in[2] = nslot (==8, compiled in)
  const float* W_proj = (const float*)d_in[3];
  const float* b_proj = (const float*)d_in[4];
  const float* ln_s   = (const float*)d_in[5];
  const float* ln_b   = (const float*)d_in[6];
  const float* W_q    = (const float*)d_in[7];
  const float* b_q    = (const float*)d_in[8];
  const float* lnq_s  = (const float*)d_in[9];
  const float* lnq_b  = (const float*)d_in[10];
  const float* W_k    = (const float*)d_in[11];
  const float* b_k    = (const float*)d_in[12];
  const float* lnk_s  = (const float*)d_in[13];
  const float* lnk_b  = (const float*)d_in[14];
  const float* W_beta = (const float*)d_in[15];
  const float* b_beta = (const float*)d_in[16];
  const float* W1     = (const float*)d_in[17];
  const float* b1     = (const float*)d_in[18];
  const float* W2     = (const float*)d_in[19];
  const float* b2     = (const float*)d_in[20];

  float* out = (float*)d_out;
  char* ws = (char*)d_ws;
  _Float16* WprojT = (_Float16*)(ws + kWsWprojT);
  _Float16* WqT    = (_Float16*)(ws + kWsWqT);
  _Float16* WkT    = (_Float16*)(ws + kWsWkT);
  _Float16* W1T    = (_Float16*)(ws + kWsW1T);
  _Float16* W2T    = (_Float16*)(ws + kWsW2T);
  float*    Mstate = (float*)(ws + kWsM);

  auto cv = [&](const float* in, _Float16* o, int K, int N) {
    int total = K * N;
    omr_convt_kernel<<<(total + 255) / 256, 256, 0, stream>>>(in, o, K, N);
  };
  cv(W_proj, WprojT, kIN, kH);        // [512,256] -> [256][512]
  cv(W_q,    WqT,    kH,  kH);
  cv(W_k,    WkT,    kH,  kH);
  cv(W1,     W1T,    2 * kH, kH4);    // [512,1024] -> [1024][512]
  cv(W2,     W2T,    kH4, kH4);       // [1024,1024] -> [1024][1024]

  // Xp projection: 512 row-tiles of 16 rows, 16 waves each
  omr_proj_kernel<<<(kT * kB) / 16, 512, 0, stream>>>(X, WprojT, b_proj, ln_s, ln_b, out);

  // Recurrent scan: 4 workgroups (one per 16-row batch tile), no grid sync needed
  omr_scan_kernel<<<4, 512, (unsigned)kScanLds, stream>>>(
      out, out, mask, WqT, WkT, W1T, W2T,
      b_q, lnq_s, lnq_b, b_k, lnk_s, lnk_b,
      W_beta, b_beta, b1, b2, ln_s, ln_b, Mstate);
}
